// AttentionHead_63290638074392
// MI455X (gfx1250) — compile-verified, matrix operand order checked by reading, not run on hardware
//
#include <hip/hip_runtime.h>

// ---------------------------------------------------------------------------
// AttentionHead for MI455X (gfx1250): B=8, S=2048, D=1024, H=64
//   k1 wconv:     W fp32 -> bf16, swizzled to WMMA B-fragment layout,
//                 k-chunk-major (12KB contiguous per K-step). Q gets the
//                 d^-0.5 * log2(e) scale folded in (softmax in log2 domain).
//   k2 proj_qkv:  fused QKV GEMM; weights staged to LDS via Tensor Data Mover
//                 (double buffered, s_wait_tensorcnt), idx streamed once (67MB,
//                 the HBM-bound part: ~3us at 23.3 TB/s). B-fragments are
//                 software double-buffered so ds_load latency hides under WMMA.
//   k3 flash_attn: causal flash attention, 64 keys/iter, bf16 WMMA scores+PV,
//                 row-sums via WMMA with an all-ones fragment, exp2 softmax.
// ---------------------------------------------------------------------------

typedef __bf16 bf16;
typedef __attribute__((ext_vector_type(16))) __bf16 v16bf;
typedef __attribute__((ext_vector_type(8)))  __bf16 v8bf;
typedef __attribute__((ext_vector_type(8)))  float  v8f;
typedef __attribute__((ext_vector_type(4)))  float  v4f;
typedef __attribute__((ext_vector_type(4))) unsigned int u32x4;
typedef __attribute__((ext_vector_type(8)))  int i32x8;
typedef __attribute__((ext_vector_type(4)))  int i32x4;

#define SEQ    2048
#define DIM    1024
#define HEAD   64
#define NROW   16384           // B*S
#define WELEMS (DIM * 192)     // swizzled combined weight elements
#define WCHUNK 6144            // elements per K-step chunk (32 x 192) = 12KB

#if __has_builtin(__builtin_amdgcn_tensor_load_to_lds)
#define HAVE_TDM 1
#else
#define HAVE_TDM 0
#endif

__device__ inline v8f wmma_bf16(v16bf a, v16bf b, v8f c) {
  return __builtin_amdgcn_wmma_f32_16x16x32_bf16(false, a, false, b, (short)0, c,
                                                 false, false);
}

// Two contiguous 16B runs -> one 16-element bf16 fragment register set.
__device__ inline v16bf load_frag(const bf16* p0, const bf16* p1) {
  v8bf a = *reinterpret_cast<const v8bf*>(p0);
  v8bf b = *reinterpret_cast<const v8bf*>(p1);
  return __builtin_shufflevector(a, b, 0, 1, 2, 3, 4, 5, 6, 7,
                                 8, 9, 10, 11, 12, 13, 14, 15);
}

__device__ inline float fast_exp2(float x) {
#if __has_builtin(__builtin_amdgcn_exp2f)
  return __builtin_amdgcn_exp2f(x);
#else
  return exp2f(x);
#endif
}

__device__ inline float red_max16(float v) {
#pragma unroll
  for (int m = 1; m < 16; m <<= 1) v = fmaxf(v, __shfl_xor(v, m, 32));
  return v;
}

#if HAVE_TDM
// TDM: 1-row tile of WCHUNK 2-byte elements, global -> LDS (D# per ISA 8.3/8.4)
__device__ inline void tdm_load_chunk(unsigned int lds_off, const void* gsrc) {
  unsigned long long ga = (unsigned long long)(uintptr_t)gsrc;
  u32x4 g0 = { 1u,                                   // count=1, user mode
               lds_off,                               // lds_addr (bytes)
               (unsigned int)ga,                      // global_addr[31:0]
               (unsigned int)((ga >> 32) & 0x1FFFFFFu) | (2u << 30) }; // type=2
  i32x8 g1 = { 0x00010000,                // wg_mask=0, data_size=1 (2 bytes)
               (int)((WCHUNK & 0xFFFF) << 16),   // tensor_dim0[15:0]
               0x00010000,                // tensor_dim0[31:16]=0, tensor_dim1=1
               (int)((WCHUNK & 0xFFFF) << 16),   // tile_dim0 = WCHUNK
               1,                         // tile_dim1=1, tile_dim2=0
               WCHUNK, 0, 0 };            // tensor_dim0_stride
  i32x4 gz = { 0, 0, 0, 0 };
#if __clang_major__ >= 23
  i32x8 gz8 = { 0, 0, 0, 0, 0, 0, 0, 0 };
  __builtin_amdgcn_tensor_load_to_lds(g0, g1, gz, gz, gz8, 0);
#else
  __builtin_amdgcn_tensor_load_to_lds(g0, g1, gz, gz, 0);
#endif
}
#else
// Fallback: cooperative 12KB copy, 48B per thread.
__device__ inline void copy_chunk(bf16* dst, const bf16* src, int tid) {
#pragma unroll
  for (int i = 0; i < 3; ++i)
    *reinterpret_cast<v8bf*>(dst + tid * 24 + i * 8) =
        *reinterpret_cast<const v8bf*>(src + tid * 24 + i * 8);
}
#endif

// ---------------------------------------------------------------------------
// Kernel 1: Wq|Wk|Wv fp32 [D,64] -> bf16 B-fragment layout, k-chunk-major:
//   frag f = k_chunk*12 + n_tile ; within frag: lane*16 + e
//   lane -> n = n_tile*16 + (lane&15)
//   e    -> k = k_chunk*32 + (lane>>4)*8 + (e&7) + 16*(e>=8)
// Wq scaled by d^-0.5 * log2(e) so attention scores land in log2 domain.
// ---------------------------------------------------------------------------
__global__ void __launch_bounds__(256) wconv(const float* __restrict__ Wk,
                                             const float* __restrict__ Wq,
                                             const float* __restrict__ Wv,
                                             bf16* __restrict__ wsW) {
  int gid  = blockIdx.x * 256 + threadIdx.x;   // 12288 threads
  int f    = gid >> 5;
  int lane = gid & 31;
  int kc   = f / 12;
  int nt   = f - kc * 12;
  int lo   = lane & 15, hi = lane >> 4;
  int n    = nt * 16 + lo;                     // 0..191: [Q | K | V]

  const float* src;
  float scale = 1.0f;
  int col;
  if (n < 64)       { src = Wq; col = n;       scale = 0.03125f * 1.44269504088896340736f; }
  else if (n < 128) { src = Wk; col = n - 64;  }
  else              { src = Wv; col = n - 128; }

  bf16* dst = wsW + (size_t)f * 512 + lane * 16;
#pragma unroll
  for (int e = 0; e < 16; ++e) {
    int k = kc * 32 + hi * 8 + (e & 7) + ((e >> 3) << 4);
    dst[e] = (bf16)(src[(size_t)k * HEAD + col] * scale);
  }
}

// ---------------------------------------------------------------------------
// Kernel 2: fused QKV projection. 8 waves x 16 rows; 12 WMMA per K-step.
// Weight chunk (12KB) staged to LDS once per WG per step via TDM (8x less L2
// traffic than per-wave loads), double-buffered. idx streamed once from HBM.
// ---------------------------------------------------------------------------
__global__ void __launch_bounds__(256) proj_qkv(const float* __restrict__ x,
                                                const bf16* __restrict__ wsW,
                                                bf16* __restrict__ Q,
                                                bf16* __restrict__ K,
                                                bf16* __restrict__ VT) {
  __shared__ bf16 wbuf[2][WCHUNK];               // 24KB double buffer
  int tid  = threadIdx.x;
  int wave = tid >> 5, lane = tid & 31;
  int lo = lane & 15, hi = lane >> 4;
  int rowbase = blockIdx.x * 128 + wave * 16;

  const float* xrow = x + (size_t)(rowbase + lo) * DIM;

  v8f z = {};
  v8f acc[12];
#pragma unroll
  for (int t = 0; t < 12; ++t) acc[t] = z;

  // Prologue: stage chunk 0
#if HAVE_TDM
  if (wave == 0) {
    tdm_load_chunk((unsigned int)(uintptr_t)&wbuf[0][0], wsW);
    __builtin_amdgcn_s_wait_tensorcnt(0);
  }
#else
  copy_chunk(&wbuf[0][0], wsW, tid);
#endif
  __syncthreads();

  for (int kc = 0; kc < 32; ++kc) {
    // Prefetch next chunk into the other buffer
#if HAVE_TDM
    if (wave == 0 && kc + 1 < 32)
      tdm_load_chunk((unsigned int)(uintptr_t)&wbuf[(kc + 1) & 1][0],
                     wsW + (size_t)(kc + 1) * WCHUNK);
#else
    if (kc + 1 < 32)
      copy_chunk(&wbuf[(kc + 1) & 1][0], wsW + (size_t)(kc + 1) * WCHUNK, tid);
#endif

    // A fragment from idx (fp32 -> bf16)
    int kk = kc * 32;
    const float* pa0 = xrow + kk + hi * 8;
    const float* pa1 = xrow + kk + 16 + hi * 8;
    v4f a0 = *(const v4f*)(pa0);
    v4f a1 = *(const v4f*)(pa0 + 4);
    v4f a2 = *(const v4f*)(pa1);
    v4f a3 = *(const v4f*)(pa1 + 4);
    v16bf A;
#pragma unroll
    for (int i = 0; i < 4; ++i) {
      A[i]      = (bf16)a0[i];
      A[4 + i]  = (bf16)a1[i];
      A[8 + i]  = (bf16)a2[i];
      A[12 + i] = (bf16)a3[i];
    }

    // B fragments: software double-buffer so the ds_load for tile t+1 is in
    // flight while the WMMA for tile t executes (breaks load->wait->wmma chain)
    const bf16* cur = &wbuf[kc & 1][0];
    v16bf Bf = load_frag(cur + lane * 16, cur + lane * 16 + 8);
#pragma unroll
    for (int t = 0; t < 12; ++t) {
      v16bf Bn = Bf;
      if (t + 1 < 12) {
        const bf16* wn = cur + (t + 1) * 512 + lane * 16;
        Bn = load_frag(wn, wn + 8);
      }
      acc[t] = wmma_bf16(A, Bf, acc[t]);
      Bf = Bn;
    }

#if HAVE_TDM
    if (wave == 0 && kc + 1 < 32) __builtin_amdgcn_s_wait_tensorcnt(0);
#endif
    __syncthreads();
  }

  // Epilogue: C layout row = rowbase + hi*8 + r, col = t*16 + lo
#pragma unroll
  for (int t = 0; t < 12; ++t) {
    int n = t * 16 + lo;
#pragma unroll
    for (int r = 0; r < 8; ++r) {
      int row = rowbase + hi * 8 + r;
      bf16 val = (bf16)acc[t][r];
      if (n < 64) {
        Q[(size_t)row * HEAD + n] = val;
      } else if (n < 128) {
        K[(size_t)row * HEAD + (n - 64)] = val;
      } else {
        int b = row >> 11, s = row & (SEQ - 1);
        VT[(size_t)b * HEAD * SEQ + (size_t)(n - 128) * SEQ + s] = val;
      }
    }
  }
}

// ---------------------------------------------------------------------------
// Kernel 3: causal flash attention, 64 keys per iteration.
//   scores: 8 WMMA ; row-sum: 2 WMMA vs all-ones ; PV: 8 WMMA
//   row-max: 16-lane xor-shuffle reduction ; probs: raw v_exp_f32 (log2 dom.)
// ---------------------------------------------------------------------------
__global__ void __launch_bounds__(256) flash_attn(const bf16* __restrict__ Q,
                                                  const bf16* __restrict__ K,
                                                  const bf16* __restrict__ VT,
                                                  float* __restrict__ out) {
  int wave = threadIdx.x >> 5, lane = threadIdx.x & 31;
  int lo = lane & 15, hi = lane >> 4;
  int b = blockIdx.x >> 4;                       // 8 batches
  int qbase = (blockIdx.x & 15) * 128 + wave * 16;

  // Q A-fragments (d^-0.5 * log2e already folded into Wq)
  const bf16* qrow = Q + ((size_t)b * SEQ + qbase + lo) * HEAD;
  v16bf qa0 = load_frag(qrow + hi * 8,      qrow + 16 + hi * 8);
  v16bf qa1 = load_frag(qrow + 32 + hi * 8, qrow + 48 + hi * 8);

  v16bf ones;
#pragma unroll
  for (int i = 0; i < 16; ++i) ones[i] = (bf16)1.0f;

  v8f z = {};
  v8f o[4];
#pragma unroll
  for (int t = 0; t < 4; ++t) o[t] = z;
  float mrow[8], lrow[8];
#pragma unroll
  for (int r = 0; r < 8; ++r) { mrow[r] = -3.0e38f; lrow[r] = 0.0f; }

  __shared__ bf16 plds[8 * 1024];                // 2KB/wave: 16x64 bf16 P tile
  bf16* pw = plds + wave * 1024;

  for (int j = 0; j <= qbase; j += 64) {
    // ---- scores: four 16x16 tiles over 64 keys
    v8f s[4];
#pragma unroll
    for (int kt = 0; kt < 4; ++kt) {
      const bf16* kp = K + ((size_t)b * SEQ + j + kt * 16 + lo) * HEAD;
      v16bf ka = load_frag(kp + hi * 8,      kp + 16 + hi * 8);
      v16bf kb = load_frag(kp + 32 + hi * 8, kp + 48 + hi * 8);
      s[kt] = wmma_bf16(qa0, ka, z);
      s[kt] = wmma_bf16(qa1, kb, s[kt]);
    }

    // ---- causal mask near the diagonal (wave-uniform branch)
    if (j + 63 > qbase) {
#pragma unroll
      for (int kt = 0; kt < 4; ++kt)
#pragma unroll
        for (int r = 0; r < 8; ++r) {
          int mg = qbase + hi * 8 + r;
          if (j + kt * 16 + lo > mg) s[kt][r] = -1.0e30f;
        }
    }

    // ---- online softmax (log2 domain): max via shuffles, probs via exp2
    float alpha[8];
#pragma unroll
    for (int r = 0; r < 8; ++r) {
      float mx = fmaxf(fmaxf(s[0][r], s[1][r]), fmaxf(s[2][r], s[3][r]));
      mx = red_max16(mx);
      float mn = fmaxf(mrow[r], mx);
      alpha[r] = fast_exp2(mrow[r] - mn);
      mrow[r] = mn;
#pragma unroll
      for (int kt = 0; kt < 4; ++kt) s[kt][r] = fast_exp2(s[kt][r] - mn);
    }

    // ---- P (C-layout) -> LDS (m*64+k) -> two A-fragments (same wave)
#pragma unroll
    for (int kt = 0; kt < 4; ++kt)
#pragma unroll
      for (int r = 0; r < 8; ++r)
        pw[(hi * 8 + r) * 64 + kt * 16 + lo] = (bf16)s[kt][r];
    v16bf pa0 = load_frag(pw + lo * 64 + hi * 8,      pw + lo * 64 + 16 + hi * 8);
    v16bf pa1 = load_frag(pw + lo * 64 + 32 + hi * 8, pw + lo * 64 + 48 + hi * 8);

    // ---- row sums via WMMA against all-ones B (replicated across lanes)
    v8f rs = wmma_bf16(pa0, ones, z);
    rs = wmma_bf16(pa1, ones, rs);
#pragma unroll
    for (int r = 0; r < 8; ++r) lrow[r] = lrow[r] * alpha[r] + rs[r];

    // ---- rescale O, then O += P x V (V^T makes B-fragments contiguous)
#pragma unroll
    for (int t = 0; t < 4; ++t)
#pragma unroll
      for (int r = 0; r < 8; ++r) o[t][r] *= alpha[r];
#pragma unroll
    for (int t = 0; t < 4; ++t) {
      const bf16* vp = VT + (size_t)b * HEAD * SEQ + (size_t)(t * 16 + lo) * SEQ + j;
      v16bf vb0 = load_frag(vp + hi * 8,      vp + 16 + hi * 8);
      v16bf vb1 = load_frag(vp + 32 + hi * 8, vp + 48 + hi * 8);
      o[t] = wmma_bf16(pa0, vb0, o[t]);
      o[t] = wmma_bf16(pa1, vb1, o[t]);
    }
  }

  // ---- normalize + write fp32 output
  float rl[8];
#pragma unroll
  for (int r = 0; r < 8; ++r) rl[r] = 1.0f / lrow[r];
#pragma unroll
  for (int t = 0; t < 4; ++t)
#pragma unroll
    for (int r = 0; r < 8; ++r)
      out[((size_t)b * SEQ + qbase + hi * 8 + r) * HEAD + t * 16 + lo] =
          o[t][r] * rl[r];
}

// ---------------------------------------------------------------------------
extern "C" void kernel_launch(void* const* d_in, const int* in_sizes, int n_in,
                              void* d_out, int out_size, void* d_ws, size_t ws_size,
                              hipStream_t stream) {
  const float* x  = (const float*)d_in[0];   // idx [8,2048,1024]
  const float* Wk = (const float*)d_in[1];   // [1024,64]
  const float* Wq = (const float*)d_in[2];
  const float* Wv = (const float*)d_in[3];
  float* out = (float*)d_out;                // [8,2048,64] fp32

  // Workspace layout (bf16): swizzled weights | Q | K | V^T  (~6.7 MB total)
  bf16* wsW = reinterpret_cast<bf16*>(d_ws);
  bf16* Q   = wsW + WELEMS;
  bf16* K   = Q + (size_t)NROW * HEAD;
  bf16* VT  = K + (size_t)NROW * HEAD;

  wconv<<<48, 256, 0, stream>>>(Wk, Wq, Wv, wsW);
  proj_qkv<<<NROW / 128, 256, 0, stream>>>(x, wsW, Q, K, VT);
  flash_attn<<<8 * (SEQ / 128), 256, 0, stream>>>(Q, K, VT, out);
}
